// Backbone_78426102825264
// MI455X (gfx1250) — compile-verified, matrix operand order checked by reading, not run on hardware
//
#include <hip/hip_runtime.h>
#include <hip/hip_bf16.h>

// ---------------------------------------------------------------------------
// Graph-transformer layer for MI455X (gfx1250, wave32, WMMA).
// Dense math: v_wmma_f32_16x16x32_f16 (f32 accumulate).
// Weight B-fragments staged in LDS (ds_load path) for the dominant GEMMs.
// ---------------------------------------------------------------------------

typedef __attribute__((ext_vector_type(8)))  _Float16 v8h;
typedef __attribute__((ext_vector_type(16))) _Float16 v16h;
typedef __attribute__((ext_vector_type(8)))  float    v8f;

static __device__ __forceinline__ v16h cat8(v8h lo, v8h hi) {
    return __builtin_shufflevector(lo, hi, 0,1,2,3,4,5,6,7,8,9,10,11,12,13,14,15);
}

// ---------------------------------------------------------------------------
// Weight pre-swizzle: fp32 row-major [K][Nout] -> f16 WMMA B-fragments.
// Fragment layout: frag[((kb*NT + nt)*32 + lane)*16 + e]
//   lane 0-15 : N = nt*16 + lane,      K in {base..base+7} U {base+16..base+23}, base=0
//   lane 16-31: N = nt*16 + (lane-16), same K pattern with base=8
//   e<8 -> K = kb*32 + base + e ; e>=8 -> K = kb*32 + base + 16 + (e-8)
// ---------------------------------------------------------------------------
__global__ void convert_weight_kernel(const float* __restrict__ W,
                                      _Float16* __restrict__ frag,
                                      int K, int Nout) {
    int tid = blockIdx.x * blockDim.x + threadIdx.x;
    int total = K * Nout;
    if (tid >= total) return;
    int NT   = Nout >> 4;
    int e    = tid & 15;
    int lane = (tid >> 4) & 31;
    int rest = tid >> 9;               // = kb*NT + nt
    int nt   = rest % NT;
    int kb   = rest / NT;
    int base = (lane < 16) ? 0 : 8;
    int k    = kb * 32 + base + (e & 7) + ((e >> 3) << 4);
    int n    = nt * 16 + (lane & 15);
    frag[tid] = (_Float16)W[(size_t)k * Nout + n];
}

// ---------------------------------------------------------------------------
// LayerNorm over 128 cols, one wave per row, 4 elems/lane.
// Optionally fuses x_dst = X + addv (stored fp32) before the LN.
// ---------------------------------------------------------------------------
__global__ void layernorm_f16_kernel(const float* __restrict__ X,
                                     const float* __restrict__ addv,   // may be null
                                     const float* __restrict__ lw,
                                     const float* __restrict__ lb,
                                     _Float16* __restrict__ Y,
                                     float* __restrict__ Xdst,         // may be null
                                     int rows) {
    int lane = threadIdx.x & 31;
    int wave = threadIdx.x >> 5;
    int row  = blockIdx.x * (blockDim.x >> 5) + wave;
    if (row >= rows) return;
    size_t rb = (size_t)row * 128;
    float4 v = *(const float4*)(X + rb + lane * 4);
    if (addv) {
        float4 a = *(const float4*)(addv + rb + lane * 4);
        v.x += a.x; v.y += a.y; v.z += a.z; v.w += a.w;
        if (Xdst) *(float4*)(Xdst + rb + lane * 4) = v;
    }
    float s  = v.x + v.y + v.z + v.w;
    float ss = v.x * v.x + v.y * v.y + v.z * v.z + v.w * v.w;
    #pragma unroll
    for (int m = 16; m > 0; m >>= 1) {
        s  += __shfl_xor(s,  m, 32);
        ss += __shfl_xor(ss, m, 32);
    }
    float mean = s * (1.0f / 128.0f);
    float var  = ss * (1.0f / 128.0f) - mean * mean;
    float rstd = rsqrtf(var + 1e-5f);
    const float* vv = &v.x;
    #pragma unroll
    for (int i = 0; i < 4; ++i) {
        int col = lane * 4 + i;
        float y = (vv[i] - mean) * rstd * lw[col] + lb[col];
        Y[rb + col] = (_Float16)y;
    }
}

// ---------------------------------------------------------------------------
// WMMA GEMM: out[rows x NT*16] = A[rows x KB*32] (f16) @ W (pre-swizzled f16)
// One wave per 16-row M-tile.
// CMODE: 0 = C := bias
//        1 = C := bias + addsrc[idx[row]][col]   (edge gather fuse)
//        2 = C := bias + addsrc[row][col]        (residual fuse)
// OMODE: 0 = store f32, 1 = store f16, 2 = relu -> f16
// STAGE: 1 = stage full B-fragment array in LDS (needs KB*NT*1024 <= 64KB)
// ---------------------------------------------------------------------------
template <int KB, int NT, int CMODE, int OMODE, int STAGE>
__global__ void gemm_wmma_kernel(const _Float16* __restrict__ A,
                                 const _Float16* __restrict__ Bfrag,
                                 const float* __restrict__ bias,
                                 const float* __restrict__ addsrc,
                                 const long long* __restrict__ idx,
                                 float* __restrict__ outF,
                                 _Float16* __restrict__ outH,
                                 int rows) {
    const int AK = KB * 32;   // A (K) width
    const int NC = NT * 16;   // output width
    __shared__ _Float16 bsh[STAGE ? (KB * NT * 512) : 32];

    if (STAGE) {
        // cooperative, coalesced copy of the swizzled weight into LDS
        const int nvec = (KB * NT * 512) / 8;   // 16B chunks
        const uint4* gsrc = (const uint4*)Bfrag;
        uint4* ldst = (uint4*)bsh;
        for (int i = threadIdx.x; i < nvec; i += blockDim.x) ldst[i] = gsrc[i];
        __syncthreads();
    }

    int lane = threadIdx.x & 31;
    int wave = threadIdx.x >> 5;
    int tile = blockIdx.x * (blockDim.x >> 5) + wave;
    if (tile * 16 >= rows) return;

    int rl    = lane & 15;
    int kbase = (lane < 16) ? 0 : 8;
    const _Float16* arow = A + (size_t)(tile * 16 + rl) * AK + kbase;
    int orow_base = tile * 16 + ((lane < 16) ? 0 : 8);

    long long gidx[8];
    if (CMODE == 1) {
        #pragma unroll
        for (int r = 0; r < 8; ++r) gidx[r] = idx[orow_base + r];
    }

    #pragma unroll 1
    for (int nt = 0; nt < NT; ++nt) {
        int col = nt * 16 + rl;
        float bv = bias[col];
        v8f c;
        #pragma unroll
        for (int r = 0; r < 8; ++r) {
            float v = bv;
            if (CMODE == 1) v += addsrc[(size_t)gidx[r] * NC + col];
            if (CMODE == 2) v += addsrc[(size_t)(orow_base + r) * NC + col];
            c[r] = v;
        }
        // k sweep in chunks of 4: issue all fragment loads, then the WMMAs,
        // so loads clause together and waits overlap compute.
        constexpr int CH = (KB >= 4) ? 4 : KB;
        #pragma unroll
        for (int kc = 0; kc < KB; kc += CH) {
            v16h areg[CH], breg[CH];
            #pragma unroll
            for (int i = 0; i < CH; ++i) {
                int kb = kc + i;
                v8h alo = *(const v8h*)(arow + kb * 32);
                v8h ahi = *(const v8h*)(arow + kb * 32 + 16);
                areg[i] = cat8(alo, ahi);
                size_t boff = (size_t)((kb * NT + nt) * 32 + lane) << 4;
                if (STAGE) {
                    v8h blo = *(const v8h*)(bsh + boff);
                    v8h bhi = *(const v8h*)(bsh + boff + 8);
                    breg[i] = cat8(blo, bhi);
                } else {
                    v8h blo = *(const v8h*)(Bfrag + boff);
                    v8h bhi = *(const v8h*)(Bfrag + boff + 8);
                    breg[i] = cat8(blo, bhi);
                }
            }
            #pragma unroll
            for (int i = 0; i < CH; ++i) {
                c = __builtin_amdgcn_wmma_f32_16x16x32_f16(
                        false, areg[i], false, breg[i], (short)0, c, false, false);
            }
        }
        #pragma unroll
        for (int r = 0; r < 8; ++r) {
            size_t o = (size_t)(orow_base + r) * NC + col;
            if (OMODE == 0) outF[o] = c[r];
            if (OMODE == 1) outH[o] = (_Float16)c[r];
            if (OMODE == 2) outH[o] = (_Float16)fmaxf(c[r], 0.0f);
        }
    }
}

// ---------------------------------------------------------------------------
// Attention pieces
// ---------------------------------------------------------------------------
__global__ void init_buffers_kernel(unsigned* __restrict__ m_u,
                                    float* __restrict__ denom,
                                    float* __restrict__ agg,
                                    int n8, int n128) {
    int tid = blockIdx.x * blockDim.x + threadIdx.x;
    if (tid < n8) { m_u[tid] = 0u; denom[tid] = 0.0f; }
    if (tid < n128) agg[tid] = 0.0f;
}

static __device__ __forceinline__ unsigned f2ord(float f) {
    unsigned u = __float_as_uint(f);
    return (u & 0x80000000u) ? ~u : (u | 0x80000000u);
}
static __device__ __forceinline__ float ord2f(unsigned u) {
    return (u & 0x80000000u) ? __uint_as_float(u ^ 0x80000000u)
                             : __uint_as_float(~u);
}

// w[e,h] = dot(q[dst[e],h,:], k_e[e,h,:]) / 4 ; atomic segment max per (dst,h)
__global__ void score_kernel(const float* __restrict__ qn,
                             const _Float16* __restrict__ ke,
                             const long long* __restrict__ dst,
                             float* __restrict__ wbuf,
                             unsigned* __restrict__ m_u,
                             int nE) {
    int tid = blockIdx.x * blockDim.x + threadIdx.x;
    if (tid >= nE * 8) return;
    int e = tid >> 3, h = tid & 7;
    long long d = dst[e];
    const float*    q = qn + (size_t)d * 128 + h * 16;
    const _Float16* k = ke + (size_t)e * 128 + h * 16;
    float s = 0.0f;
    #pragma unroll
    for (int i = 0; i < 16; ++i) s += q[i] * (float)k[i];
    s *= 0.25f;  // 1/sqrt(16)
    wbuf[tid] = s;
    atomicMax(&m_u[(size_t)d * 8 + h], f2ord(s));
}

__global__ void expsum_kernel(const long long* __restrict__ dst,
                              const unsigned* __restrict__ m_u,
                              float* __restrict__ wbuf,
                              float* __restrict__ denom,
                              int nE) {
    int tid = blockIdx.x * blockDim.x + threadIdx.x;
    if (tid >= nE * 8) return;
    int e = tid >> 3, h = tid & 7;
    long long d = dst[e];
    float mf = ord2f(m_u[(size_t)d * 8 + h]);
    float ew = expf(wbuf[tid] - mf);
    wbuf[tid] = ew;
    atomicAdd(&denom[(size_t)d * 8 + h], ew);
}

// agg[dst[e], c] += alpha[e, c/16] * v_e[e, c]
__global__ void aggregate_kernel(const long long* __restrict__ dst,
                                 const float* __restrict__ wbuf,
                                 const float* __restrict__ denom,
                                 const _Float16* __restrict__ ve,
                                 float* __restrict__ agg,
                                 long long total) {
    long long tid = (long long)blockIdx.x * blockDim.x + threadIdx.x;
    if (tid >= total) return;
    int e = (int)(tid >> 7);
    int c = (int)(tid & 127);
    int h = c >> 4;
    long long d = dst[e];
    float alpha = wbuf[(size_t)e * 8 + h] / (denom[(size_t)d * 8 + h] + 1e-16f);
    atomicAdd(&agg[(size_t)d * 128 + c], alpha * (float)ve[tid]);
}

// ---------------------------------------------------------------------------
// Host-side launch
// ---------------------------------------------------------------------------
extern "C" void kernel_launch(void* const* d_in, const int* in_sizes, int n_in,
                              void* d_out, int out_size, void* d_ws, size_t ws_size,
                              hipStream_t stream) {
    (void)in_sizes; (void)n_in; (void)out_size; (void)ws_size;
    const int N = 50000, E = 800000;

    const float*     x    = (const float*)d_in[0];
    const long long* eidx = (const long long*)d_in[1];
    const float*     eatt = (const float*)d_in[2];
    const float *Wq = (const float*)d_in[3],  *bq  = (const float*)d_in[4];
    const float *Wk = (const float*)d_in[5],  *bk  = (const float*)d_in[6];
    const float *Wv = (const float*)d_in[7],  *bv  = (const float*)d_in[8];
    const float *Wek= (const float*)d_in[9],  *bek = (const float*)d_in[10];
    const float *Wev= (const float*)d_in[11], *bev = (const float*)d_in[12];
    const float *W1 = (const float*)d_in[13], *b1  = (const float*)d_in[14];
    const float *W2 = (const float*)d_in[15], *b2  = (const float*)d_in[16];
    const float *lnsw = (const float*)d_in[17], *lnsb = (const float*)d_in[18];
    const float *lnew = (const float*)d_in[19], *lneb = (const float*)d_in[20];
    const float *lnfw = (const float*)d_in[21], *lnfb = (const float*)d_in[22];
    float* out = (float*)d_out;

    const long long* src = eidx;       // edge_index[0]
    const long long* dst = eidx + E;   // edge_index[1]

    // ---- workspace carve-out (256B aligned) ----
    char* ws = (char*)d_ws;
    size_t off = 0;
    auto alloc = [&](size_t bytes) -> char* {
        char* p = ws + off;
        off = (off + bytes + 255) & ~(size_t)255;
        return p;
    };
    _Float16* xs16  = (_Float16*)alloc((size_t)N * 128 * 2);
    _Float16* ea16  = (_Float16*)alloc((size_t)E * 128 * 2);
    float*    qn    = (float*)   alloc((size_t)N * 128 * 4);
    float*    kn    = (float*)   alloc((size_t)N * 128 * 4);
    float*    vn    = (float*)   alloc((size_t)N * 128 * 4);
    _Float16* k_e   = (_Float16*)alloc((size_t)E * 128 * 2);
    _Float16* v_e   = (_Float16*)alloc((size_t)E * 128 * 2);
    float*    wbuf  = (float*)   alloc((size_t)E * 8 * 4);
    unsigned* m_u   = (unsigned*)alloc((size_t)N * 8 * 4);
    float*    denom = (float*)   alloc((size_t)N * 8 * 4);
    float*    agg   = (float*)   alloc((size_t)N * 128 * 4);
    float*    x_dst = (float*)   alloc((size_t)N * 128 * 4);
    _Float16* h16   = (_Float16*)alloc((size_t)N * 128 * 2);
    _Float16* hid16 = (_Float16*)alloc((size_t)N * 512 * 2);
    _Float16* WqF   = (_Float16*)alloc(128 * 128 * 2);
    _Float16* WkF   = (_Float16*)alloc(128 * 128 * 2);
    _Float16* WvF   = (_Float16*)alloc(128 * 128 * 2);
    _Float16* WekF  = (_Float16*)alloc(128 * 128 * 2);
    _Float16* WevF  = (_Float16*)alloc(128 * 128 * 2);
    _Float16* W1F   = (_Float16*)alloc(128 * 512 * 2);
    _Float16* W2F   = (_Float16*)alloc(512 * 128 * 2);

    // ---- 1. weight swizzle (fp32 -> f16 WMMA fragments) ----
    convert_weight_kernel<<<(128*128 + 255) / 256, 256, 0, stream>>>(Wq,  WqF,  128, 128);
    convert_weight_kernel<<<(128*128 + 255) / 256, 256, 0, stream>>>(Wk,  WkF,  128, 128);
    convert_weight_kernel<<<(128*128 + 255) / 256, 256, 0, stream>>>(Wv,  WvF,  128, 128);
    convert_weight_kernel<<<(128*128 + 255) / 256, 256, 0, stream>>>(Wek, WekF, 128, 128);
    convert_weight_kernel<<<(128*128 + 255) / 256, 256, 0, stream>>>(Wev, WevF, 128, 128);
    convert_weight_kernel<<<(128*512 + 255) / 256, 256, 0, stream>>>(W1,  W1F,  128, 512);
    convert_weight_kernel<<<(512*128 + 255) / 256, 256, 0, stream>>>(W2,  W2F,  512, 128);

    // ---- 2. layernorms -> f16 A operands ----
    layernorm_f16_kernel<<<(N + 7) / 8, 256, 0, stream>>>(x, nullptr, lnsw, lnsb, xs16, nullptr, N);
    layernorm_f16_kernel<<<(E + 7) / 8, 256, 0, stream>>>(eatt, nullptr, lnew, lneb, ea16, nullptr, E);

    // ---- 3. node projections: qn/kn/vn = xs @ W + b (f32 out), LDS-staged B ----
    int ngrid16 = (N / 16 + 15) / 16;   // 512-thread blocks, 16 tiles each
    gemm_wmma_kernel<4, 8, 0, 0, 1><<<ngrid16, 512, 0, stream>>>(xs16, WqF, bq, nullptr, nullptr, qn, nullptr, N);
    gemm_wmma_kernel<4, 8, 0, 0, 1><<<ngrid16, 512, 0, stream>>>(xs16, WkF, bk, nullptr, nullptr, kn, nullptr, N);
    gemm_wmma_kernel<4, 8, 0, 0, 1><<<ngrid16, 512, 0, stream>>>(xs16, WvF, bv, nullptr, nullptr, vn, nullptr, N);

    // ---- 4. edge projections fused with gather: k_e = ea@Wek + bek + kn[src] ----
    int egrid16 = (E / 16 + 15) / 16;
    gemm_wmma_kernel<4, 8, 1, 1, 1><<<egrid16, 512, 0, stream>>>(ea16, WekF, bek, kn, src, nullptr, k_e, E);
    gemm_wmma_kernel<4, 8, 1, 1, 1><<<egrid16, 512, 0, stream>>>(ea16, WevF, bev, vn, src, nullptr, v_e, E);

    // ---- 5. segment softmax + scatter aggregate ----
    init_buffers_kernel<<<(N * 128 + 255) / 256, 256, 0, stream>>>(m_u, denom, agg, N * 8, N * 128);
    score_kernel<<<(E * 8 + 255) / 256, 256, 0, stream>>>(qn, k_e, dst, wbuf, m_u, E);
    expsum_kernel<<<(E * 8 + 255) / 256, 256, 0, stream>>>(dst, m_u, wbuf, denom, E);
    long long atot = (long long)E * 128;
    aggregate_kernel<<<(unsigned)((atot + 255) / 256), 256, 0, stream>>>(dst, wbuf, denom, v_e, agg, atot);

    // ---- 6. residual + LN -> FFN (weights too large for single LDS stage) ----
    int ngrid8 = (N / 16 + 7) / 8;
    layernorm_f16_kernel<<<(N + 7) / 8, 256, 0, stream>>>(x, agg, lnfw, lnfb, h16, x_dst, N);
    gemm_wmma_kernel<4, 32, 0, 2, 0><<<ngrid8, 256, 0, stream>>>(h16, W1F, b1, nullptr, nullptr, nullptr, hid16, N);
    gemm_wmma_kernel<16, 8, 2, 0, 0><<<ngrid8, 256, 0, stream>>>(hid16, W2F, b2, x_dst, nullptr, out, nullptr, N);
}
